// Q_ConvSeq3x3Branch_88699664597376
// MI455X (gfx1250) — 8-bit matrix operand order repaired and checked by reading, compiled, not run on hardware
//
#include <hip/hip_runtime.h>
#include <stdint.h>

#define QN_LVL 127.0f
#define Q_BN_EPS 1e-5f

typedef __attribute__((ext_vector_type(8))) int v8i;

__global__ void init_amax_kernel(unsigned* amax) {
  if (threadIdx.x < 4) amax[threadIdx.x] = 0u;
}

__global__ void absmax_f32_kernel(const float* __restrict__ x, size_t n,
                                  unsigned* __restrict__ amax) {
  __shared__ float red[256];
  float lm = 0.f;
  for (size_t i = (size_t)blockIdx.x * blockDim.x + threadIdx.x; i < n;
       i += (size_t)gridDim.x * blockDim.x)
    lm = fmaxf(lm, fabsf(x[i]));
  red[threadIdx.x] = lm;
  __syncthreads();
  for (int s = 128; s > 0; s >>= 1) {
    if ((int)threadIdx.x < s) red[threadIdx.x] = fmaxf(red[threadIdx.x], red[threadIdx.x + s]);
    __syncthreads();
  }
  if (threadIdx.x == 0) atomicMax(amax, __float_as_uint(red[0]));
}

__global__ void fold_quant_w_kernel(const float* __restrict__ w, const float* __restrict__ g,
                                    const float* __restrict__ b, const float* __restrict__ mu,
                                    const float* __restrict__ var,
                                    const unsigned* __restrict__ amax_act,
                                    int8_t* __restrict__ wq, float* __restrict__ bint,
                                    float* __restrict__ csf, int Cin, int KH, int KW) {
  const int oc = blockIdx.x;
  const int K = Cin * KH * KW;
  const float stdv = sqrtf(var[oc] + Q_BN_EPS);
  const float sc = g[oc] / stdv;
  const float* wrow = w + (size_t)oc * K;
  __shared__ float red[256];
  __shared__ float s_wsf;
  float lm = 0.f;
  for (int e = threadIdx.x; e < K; e += 256) lm = fmaxf(lm, fabsf(wrow[e] * sc));
  red[threadIdx.x] = lm;
  __syncthreads();
  for (int s = 128; s > 0; s >>= 1) {
    if ((int)threadIdx.x < s) red[threadIdx.x] = fmaxf(red[threadIdx.x], red[threadIdx.x + s]);
    __syncthreads();
  }
  if (threadIdx.x == 0) {
    float wsf = red[0] / QN_LVL;
    s_wsf = wsf;
    float a_sf = __uint_as_float(*amax_act) / QN_LVL;
    float cs = a_sf * wsf;
    csf[oc] = cs;
    float bf = b[oc] - g[oc] * mu[oc] / stdv;
    bint[oc] = rintf(bf / cs);
  }
  __syncthreads();
  const float inv_wsf = 1.f / s_wsf;
  int8_t* wdst = wq + (size_t)oc * K;
  const int khw = KH * KW;
  for (int e = threadIdx.x; e < K; e += 256) {
    int ci = e / khw;
    int r = e - ci * khw;
    int khi = r / KW;
    int kwi = r - khi * KW;
    float q = rintf(wrow[e] * sc * inv_wsf);
    q = fminf(fmaxf(q, -128.f), 127.f);
    wdst[(size_t)(khi * KW + kwi) * Cin + ci] = (int8_t)(int)q;
  }
}

__global__ void quant_act_i8_kernel(const float* __restrict__ y, const unsigned* __restrict__ amax,
                                    int8_t* __restrict__ xq, size_t n, int C, int HW,
                                    int from_nchw) {
  const float s = __uint_as_float(*amax) / QN_LVL;
  const float inv = 1.f / s;
  for (size_t i = (size_t)blockIdx.x * blockDim.x + threadIdx.x; i < n;
       i += (size_t)gridDim.x * blockDim.x) {
    float q = rintf(y[i] * inv);
    q = fminf(fmaxf(q, -128.f), 127.f);
    size_t d = i;
    if (from_nchw) {
      size_t chw = (size_t)C * HW;
      size_t img = i / chw;
      size_t r = i - img * chw;
      size_t c = r / HW;
      size_t p = r - c * HW;
      d = (img * (size_t)HW + p) * C + c;
    }
    xq[d] = (int8_t)(int)q;
  }
}

__global__ __launch_bounds__(256) void conv_wmma_iu8_kernel(
    const int8_t* __restrict__ xq, const int8_t* __restrict__ wq,
    const float* __restrict__ bint, const float* __restrict__ csf, float* __restrict__ y,
    unsigned* __restrict__ amax, int H, int W, int Cin, int Cout, int KH, int KW, int PH, int PW,
    int ostride, int ooff, int nchw_out) {
  const int K = Cin * KH * KW;
  const int HWp = H * W;
  __shared__ int As[2][64 * 17];
  __shared__ int Bs[2][64 * 17];
  __shared__ float red[256];
  const int tid = threadIdx.x;
  const int lane = tid & 31;
  const int wv = tid >> 5;
  const int mTile = (wv >> 1) * 16;
  const int nTile = (wv & 1) * 32;
  const int m0 = blockIdx.x * 64;
  const int n0 = blockIdx.y * 64;

  const int half = lane >> 4;
  const int aRow = mTile + (lane & 15);
  const int bRow0 = nTile + (lane & 15);
  const int bRow1 = bRow0 + 16;
  const int aKhalf = half * 8;
  const int bKhalf = half * 16;

  int a_ml[4], a_kb[4], a_img[4], a_oh[4], a_ow[4];
#pragma unroll
  for (int t = 0; t < 4; ++t) {
    int wi = tid + t * 256;
    a_ml[t] = wi >> 4;
    a_kb[t] = (wi & 15) * 4;
    int m = m0 + a_ml[t];
    int img = m / HWp;
    int rem = m - img * HWp;
    a_img[t] = img;
    a_oh[t] = rem / W;
    a_ow[t] = rem - a_oh[t] * W;
  }

  auto load_tiles = [&](int k0s, int bsel) {
#pragma unroll
    for (int t = 0; t < 4; ++t) {
      int kk = k0s + a_kb[t];
      int seg = kk / Cin;
      int ci = kk - seg * Cin;
      int khi = seg / KW;
      int kwi = seg - khi * KW;
      int ih = a_oh[t] - PH + khi;
      int iw = a_ow[t] - PW + kwi;
      int val = 0;
      if ((unsigned)ih < (unsigned)H && (unsigned)iw < (unsigned)W)
        val = *(const int*)(xq + ((size_t)((a_img[t] * H + ih) * W + iw) * Cin + ci));
      As[bsel][a_ml[t] * 17 + (a_kb[t] >> 2)] = val;
    }
#pragma unroll
    for (int t = 0; t < 4; ++t) {
      int wi = tid + t * 256;
      int nl = wi >> 4;
      int kb = (wi & 15) * 4;
      Bs[bsel][nl * 17 + (kb >> 2)] = *(const int*)(wq + ((size_t)(n0 + nl) * K + k0s + kb));
    }
  };

  v8i acc0 = {0, 0, 0, 0, 0, 0, 0, 0};
  v8i acc1 = {0, 0, 0, 0, 0, 0, 0, 0};

  load_tiles(0, 0);
  int buf = 0;
  for (int k0 = 0; k0 < K; k0 += 64) {
    __syncthreads();
    if (k0 + 64 < K) load_tiles(k0 + 64, buf ^ 1);

    v8i a, b0, b1;
#pragma unroll
    for (int v = 0; v < 8; ++v) {
      int ka = ((v & 3) * 4 + (v >> 2) * 32 + bKhalf) >> 2;
      a[v] = As[buf][aRow * 17 + ka];
      int kb = ((v & 3) * 4 + (v >> 2) * 32 + bKhalf) >> 2;
      b0[v] = Bs[buf][bRow0 * 17 + kb];
      b1[v] = Bs[buf][bRow1 * 17 + kb];
    }
    acc0 = __builtin_amdgcn_wmma_i32_16x16x64_iu8(true, a, true, b0, acc0, false, false);
    acc1 = __builtin_amdgcn_wmma_i32_16x16x64_iu8(true, a, true, b1, acc1, false, false);
    buf ^= 1;
  }

  const int oc0 = n0 + nTile + (lane & 15);
  const int oc1 = oc0 + 16;
  const float cs0 = csf[oc0], bi0 = bint[oc0];
  const float cs1 = csf[oc1], bi1 = bint[oc1];
  float lmax = 0.f;
#pragma unroll
  for (int v = 0; v < 8; ++v) {
    int ml = mTile + v + half * 8;
    int m = m0 + ml;
    int img = m / HWp;
    int rem = m - img * HWp;
    float f0 = fmaxf(((float)acc0[v] + bi0) * cs0, 0.f);
    float f1 = fmaxf(((float)acc1[v] + bi1) * cs1, 0.f);
    lmax = fmaxf(lmax, fmaxf(f0, f1));
    if (nchw_out) {
      y[((size_t)img * ostride + (ooff + oc0)) * (size_t)HWp + rem] = f0;
      y[((size_t)img * ostride + (ooff + oc1)) * (size_t)HWp + rem] = f1;
    } else {
      y[(size_t)m * ostride + ooff + oc0] = f0;
      y[(size_t)m * ostride + ooff + oc1] = f1;
    }
  }
  red[tid] = lmax;
  __syncthreads();
  for (int s = 128; s > 0; s >>= 1) {
    if (tid < s) red[tid] = fmaxf(red[tid], red[tid + s]);
    __syncthreads();
  }
  if (tid == 0) atomicMax(amax, __float_as_uint(red[0]));
}

__global__ void final_requant_kernel(float* __restrict__ out, const unsigned* __restrict__ amax,
                                     size_t n) {
  const float s = __uint_as_float(*amax) / QN_LVL;
  const float inv = 1.f / s;
  for (size_t i = (size_t)blockIdx.x * blockDim.x + threadIdx.x; i < n;
       i += (size_t)gridDim.x * blockDim.x) {
    float q = rintf(out[i] * inv);
    q = fminf(fmaxf(q, -128.f), 127.f);
    out[i] = q * s;
  }
}

extern "C" void kernel_launch(void* const* d_in, const int* in_sizes, int n_in, void* d_out,
                              int out_size, void* d_ws, size_t ws_size, hipStream_t stream) {
  (void)in_sizes; (void)n_in; (void)ws_size;
  const float* x  = (const float*)d_in[0];
  const float* w1 = (const float*)d_in[1];  const float* g1 = (const float*)d_in[2];
  const float* b1 = (const float*)d_in[3];  const float* m1 = (const float*)d_in[4];
  const float* v1 = (const float*)d_in[5];
  const float* w2 = (const float*)d_in[6];  const float* g2 = (const float*)d_in[7];
  const float* b2 = (const float*)d_in[8];  const float* m2 = (const float*)d_in[9];
  const float* v2 = (const float*)d_in[10];
  const float* w3 = (const float*)d_in[11]; const float* g3 = (const float*)d_in[12];
  const float* b3 = (const float*)d_in[13]; const float* m3 = (const float*)d_in[14];
  const float* v3 = (const float*)d_in[15];
  const float* w4 = (const float*)d_in[16]; const float* g4 = (const float*)d_in[17];
  const float* b4 = (const float*)d_in[18]; const float* m4 = (const float*)d_in[19];
  const float* v4 = (const float*)d_in[20];

  const int Nimg = 64, H = 17, W = 17, HW = H * W;
  const size_t M = (size_t)Nimg * HW;

  char* ws = (char*)d_ws;
  size_t off = 0;
  auto alloc = [&](size_t bytes) -> void* {
    off = (off + 255) & ~(size_t)255;
    void* p = ws + off;
    off += bytes;
    return p;
  };
  unsigned* amax = (unsigned*)alloc(4 * sizeof(unsigned));
  float* bint1 = (float*)alloc(448 * 4); float* csf1 = (float*)alloc(448 * 4);
  float* bint2 = (float*)alloc(384 * 4); float* csf2 = (float*)alloc(384 * 4);
  float* bint3 = (float*)alloc(384 * 4); float* csf3 = (float*)alloc(384 * 4);
  float* bint4 = (float*)alloc(384 * 4); float* csf4 = (float*)alloc(384 * 4);
  int8_t* wq1 = (int8_t*)alloc((size_t)448 * 384);
  int8_t* wq2 = (int8_t*)alloc((size_t)384 * 4032);
  int8_t* wq3 = (int8_t*)alloc((size_t)384 * 1152);
  int8_t* wq4 = (int8_t*)alloc((size_t)384 * 1152);
  int8_t* xq0 = (int8_t*)alloc(M * 384);
  int8_t* xq1 = (int8_t*)alloc(M * 448);
  int8_t* xq2 = (int8_t*)alloc(M * 384);
  float* yS = (float*)alloc(M * 448 * 4);

  float* out = (float*)d_out;

  init_amax_kernel<<<1, 32, 0, stream>>>(amax);

  const size_t nx = M * 384;
  absmax_f32_kernel<<<1024, 256, 0, stream>>>(x, nx, amax + 0);
  fold_quant_w_kernel<<<448, 256, 0, stream>>>(w1, g1, b1, m1, v1, amax + 0, wq1, bint1, csf1,
                                               384, 1, 1);
  quant_act_i8_kernel<<<2048, 256, 0, stream>>>(x, amax + 0, xq0, nx, 384, HW, 1);

  conv_wmma_iu8_kernel<<<dim3(289, 7), 256, 0, stream>>>(xq0, wq1, bint1, csf1, yS, amax + 1,
                                                         H, W, 384, 448, 1, 1, 0, 0, 448, 0, 0);
  fold_quant_w_kernel<<<384, 256, 0, stream>>>(w2, g2, b2, m2, v2, amax + 1, wq2, bint2, csf2,
                                               448, 3, 3);
  quant_act_i8_kernel<<<2048, 256, 0, stream>>>(yS, amax + 1, xq1, M * 448, 448, HW, 0);

  conv_wmma_iu8_kernel<<<dim3(289, 6), 256, 0, stream>>>(xq1, wq2, bint2, csf2, yS, amax + 2,
                                                         H, W, 448, 384, 3, 3, 1, 1, 384, 0, 0);
  fold_quant_w_kernel<<<384, 256, 0, stream>>>(w3, g3, b3, m3, v3, amax + 2, wq3, bint3, csf3,
                                               384, 1, 3);
  fold_quant_w_kernel<<<384, 256, 0, stream>>>(w4, g4, b4, m4, v4, amax + 2, wq4, bint4, csf4,
                                               384, 3, 1);
  quant_act_i8_kernel<<<2048, 256, 0, stream>>>(yS, amax + 2, xq2, M * 384, 384, HW, 0);

  conv_wmma_iu8_kernel<<<dim3(289, 6), 256, 0, stream>>>(xq2, wq3, bint3, csf3, out, amax + 3,
                                                         H, W, 384, 384, 1, 3, 0, 1, 768, 0, 1);
  conv_wmma_iu8_kernel<<<dim3(289, 6), 256, 0, stream>>>(xq2, wq4, bint4, csf4, out, amax + 3,
                                                         H, W, 384, 384, 3, 1, 1, 0, 768, 384, 1);

  final_requant_kernel<<<2048, 256, 0, stream>>>(out, amax + 3, (size_t)out_size);
}
